// GATDiscriminator_19499151524162
// MI455X (gfx1250) — matrix operand order repaired and checked by reading, not compiled, not run on hardware
//
#include <hip/hip_runtime.h>
#include <hip/hip_bf16.h>

// ---------------------------------------------------------------------------
// GAT discriminator for MI455X (gfx1250, wave32).
//   layer: h = x@W (bf16 WMMA, f32 accum; W staged via TDM, A stripe staged
//          via async global->LDS) -> per-head attn scores ->
//          edge softmax (CDNA5 float atomics) -> scatter aggregate -> tanh
// ---------------------------------------------------------------------------

typedef __attribute__((ext_vector_type(16))) __bf16       v16bf;
typedef __attribute__((ext_vector_type(8)))  float        v8f;
typedef __attribute__((ext_vector_type(4)))  unsigned int u32x4;
typedef __attribute__((ext_vector_type(8)))  int          i32x8;
typedef __attribute__((ext_vector_type(4)))  int          i32x4;

// Address-space-qualified vector pointers for the async global->LDS builtin.
typedef __attribute__((address_space(1))) i32x4 as1_i32x4;   // global
typedef __attribute__((address_space(3))) i32x4 as3_i32x4;   // LDS

#define EMB    128
#define HF     128
#define HEADS  8
#define HID    16
#define NEG_SLOPE 0.2f

// ---------------- CDNA5 no-return float atomics (device scope) -------------
__device__ __forceinline__ void atomic_add_f32(float* p, float v) {
  asm volatile("global_atomic_add_f32 %0, %1, off scope:SCOPE_DEV"
               :: "v"(p), "v"(v) : "memory");
}
__device__ __forceinline__ void atomic_max_f32(float* p, float v) {
  asm volatile("global_atomic_max_num_f32 %0, %1, off scope:SCOPE_DEV"
               :: "v"(p), "v"(v) : "memory");
}

// LDS byte offset of a generic pointer known to be in shared memory.
__device__ __forceinline__ unsigned lds_offset_of(const void* p) {
  return (unsigned)(size_t)(__attribute__((address_space(3))) const void*)p;
}

// ---------------- f32 -> bf16 conversions ----------------------------------
__global__ void cvt_f32_bf16(const float* __restrict__ in,
                             __bf16* __restrict__ out, int n) {
  int i = blockIdx.x * blockDim.x + threadIdx.x;
  if (i < n) out[i] = (__bf16)in[i];
}

// W is [K=128, N=128] row-major f32; produce WT bf16 [N, K] row-major.
__global__ void transpose_cvt_w(const float* __restrict__ W,
                                __bf16* __restrict__ wt) {
  int idx = blockIdx.x * blockDim.x + threadIdx.x;
  if (idx >= HF * HF) return;
  int n = idx >> 7, k = idx & 127;
  wt[idx] = (__bf16)W[k * HF + n];
}

// ---------------- GEMM: h[M,128] = xb[M,128] @ WT[128,128]^T ---------------
// Block = 256 threads = 8 waves. Block stages its 16x128 A stripe (4 KB,
// async global->LDS) and the whole 128x128 bf16 W (32 KB, Tensor Data Mover)
// into LDS; each wave then computes one 16x16 output tile with
// 4x v_wmma_f32_16x16x32_bf16 fed by ds_load_b128 fragments.
__global__ void __launch_bounds__(256)
gemm_xw_wmma(const __bf16* __restrict__ xb,   // [M,128] bf16 row-major
             const __bf16* __restrict__ wt,   // [128(N),128(K)] bf16 row-major
             float* __restrict__ h,           // [M,128] f32
             int M) {
  __shared__ __bf16 sW[HF * HF];    // 32 KB: whole W (transposed layout)
  __shared__ __bf16 sA[16 * EMB];   // 4 KB: this block's A stripe

  const int tid   = threadIdx.x;
  const int lane  = tid & 31;
  const int wv    = tid >> 5;          // n-tile 0..7
  const int mbase = blockIdx.x * 16;
  const int nbase = wv * 16;
  const int l16   = lane & 15;
  const int khalf = lane >> 4;

  // ---- stage A stripe: each thread moves one 16-byte chunk ---------------
  {
    int row  = tid >> 4;               // 0..15
    int seg  = tid & 15;               // 16-byte chunk within the row
    int mrow = mbase + row;
    if (mrow >= M) mrow = M - 1;       // clamp (M=50000 is exact)
    const __bf16* gsrc = xb + (size_t)mrow * EMB + seg * 8;
    __bf16*       ldst = sA + row * EMB + seg * 8;
#if __has_builtin(__builtin_amdgcn_global_load_async_to_lds_b128)
    // Direct global->LDS copy, no VGPR round trip (ASYNCcnt tracked).
    __builtin_amdgcn_global_load_async_to_lds_b128(
        (as1_i32x4*)(void*)gsrc, (as3_i32x4*)(void*)ldst, 0, 0);
#if __has_builtin(__builtin_amdgcn_s_wait_asynccnt)
    __builtin_amdgcn_s_wait_asynccnt(0);
#else
    asm volatile("s_wait_asynccnt 0" ::: "memory");
#endif
#else
    *reinterpret_cast<float4*>(ldst) =
        *reinterpret_cast<const float4*>(gsrc);
#endif
  }

  // ---- stage W via TDM (wave 0 issues one tensor_load_to_lds) ------------
#if __has_builtin(__builtin_amdgcn_tensor_load_to_lds) && \
    __has_builtin(__builtin_amdgcn_s_wait_tensorcnt)
  if (wv == 0) {
    const unsigned long long ga = (unsigned long long)wt;
    // D# group 0: count=1 | lds_addr | global_addr[56:0] | type=2
    u32x4 g0;
    g0[0] = 1u;                                    // count=1, user mode
    g0[1] = lds_offset_of(sW);                     // lds_addr (bytes)
    g0[2] = (unsigned)ga;                          // global_addr[31:0]
    g0[3] = ((unsigned)(ga >> 32) & 0x01FFFFFFu)   // global_addr[56:32]
            | (2u << 30);                          // type = "image"
    // D# group 1: data_size=2B, tensor 128x128, tile 128x128, stride0=128
    i32x8 g1;
    g1[0] = 0x00010000;        // workgroup_mask=0, data_size=1 (2 bytes)
    g1[1] = (int)(128u << 16); // tensor_dim0[15:0] = 128
    g1[2] = (int)(128u << 16); // tensor_dim0 hi=0 | tensor_dim1[15:0]=128
    g1[3] = (int)(128u << 16); // tensor_dim1 hi=0 | tile_dim0=128
    g1[4] = 128;               // tile_dim1=128, tile_dim2=0
    g1[5] = 128;               // tensor_dim0_stride[31:0] = 128
    g1[6] = 0;                 // stride0 hi | tensor_dim1_stride lo
    g1[7] = 0;
    i32x4 z4 = {0, 0, 0, 0};             // groups 2/3 unused (2-D tensor)
    i32x8 z8 = {0, 0, 0, 0, 0, 0, 0, 0}; // extra group (clang-23 6-arg form)
    __builtin_amdgcn_tensor_load_to_lds(g0, g1, z4, z4, z8, 0);
    __builtin_amdgcn_s_wait_tensorcnt((short)0);
  }
#else
  for (int i = tid * 8; i < HF * HF; i += 256 * 8)
    *reinterpret_cast<float4*>(sW + i) =
        *reinterpret_cast<const float4*>(wt + i);
#endif
  __syncthreads();

  // ---- WMMA main loop (fragments from LDS) -------------------------------
  const __bf16* arow = sA + l16 * EMB;
  const __bf16* brow = sW + (nbase + l16) * EMB;

  v8f c = {};
#pragma unroll
  for (int kk = 0; kk < 4; ++kk) {
    const int k0 = kk * 32;
    v16bf a, b;
    // A fragment, ISA 16-bit 16x32 layout: lane holds two 8-element K runs.
    *reinterpret_cast<float4*>(&a) =
        *reinterpret_cast<const float4*>(arow + k0 + khalf * 8);
    *(reinterpret_cast<float4*>(&a) + 1) =
        *reinterpret_cast<const float4*>(arow + k0 + 16 + khalf * 8);
    // B fragment: lane (n = lane&15) holds two 8-element K runs, same as A.
    *reinterpret_cast<float4*>(&b) =
        *reinterpret_cast<const float4*>(brow + k0 + khalf * 8);
    *(reinterpret_cast<float4*>(&b) + 1) =
        *reinterpret_cast<const float4*>(brow + k0 + 16 + khalf * 8);

    c = __builtin_amdgcn_wmma_f32_16x16x32_bf16(
        /*neg_a=*/false, a, /*neg_b=*/false, b,
        /*c_mod=*/(short)0, c, /*reuse_a=*/false, /*reuse_b=*/false);
  }

  // C/D layout: VGPR r -> rows (khalf*8 + r), cols = lane&15.
  const int col   = nbase + l16;
  const int rbase = mbase + (khalf << 3);
  if (mbase + 16 <= M) {                 // uniform fast path (always for 50000)
#pragma unroll
    for (int r = 0; r < 8; ++r)
      h[(size_t)(rbase + r) * HF + col] = c[r];
  } else {
#pragma unroll
    for (int r = 0; r < 8; ++r)
      if (rbase + r < M) h[(size_t)(rbase + r) * HF + col] = c[r];
  }
}

// ---------------- per-node attention scores --------------------------------
__global__ void node_scores(const float* __restrict__ h,
                            const float* __restrict__ a_src,
                            const float* __restrict__ a_dst,
                            float* __restrict__ ssrc,
                            float* __restrict__ sdst, int N) {
  int idx = blockIdx.x * blockDim.x + threadIdx.x;
  if (idx >= N * HEADS) return;
  int head = idx & 7, n = idx >> 3;
  const float* hp = h + (size_t)n * HF + head * HID;
  const float* as = a_src + head * HID;
  const float* ad = a_dst + head * HID;
  float s1 = 0.f, s2 = 0.f;
#pragma unroll
  for (int f = 0; f < HID; ++f) {
    float v = hp[f];
    s1 += v * as[f];
    s2 += v * ad[f];
  }
  ssrc[idx] = s1;
  sdst[idx] = s2;
}

// ---------------- init m/denom/agg -----------------------------------------
__global__ void init_layer(float* __restrict__ agg, float* __restrict__ m,
                           float* __restrict__ denom, int N) {
  int idx = blockIdx.x * blockDim.x + threadIdx.x;
  if (idx < N * HF) agg[idx] = 0.f;
  if (idx < N * HEADS) { m[idx] = -3.0e38f; denom[idx] = 0.f; }
}

// ---------------- edge pass 1: logits + segment max ------------------------
__global__ void edge_logits(const long long* __restrict__ ei, int E0, int ET,
                            const float* __restrict__ ssrc,
                            const float* __restrict__ sdst,
                            float* __restrict__ e, float* __restrict__ m) {
  int idx = blockIdx.x * blockDim.x + threadIdx.x;
  if (idx >= ET * HEADS) return;
  int edge = idx >> 3, head = idx & 7;
  long long s, d;
  if (edge < E0) { s = ei[edge]; d = ei[(size_t)E0 + edge]; }
  else           { s = d = (long long)(edge - E0); }
  float v = ssrc[s * HEADS + head] + sdst[d * HEADS + head];
  v = v > 0.f ? v : NEG_SLOPE * v;                   // leaky relu
  e[idx] = v;
  atomic_max_f32(&m[d * HEADS + head], v);
}

// ---------------- edge pass 2: exp + segment sum ---------------------------
__global__ void edge_exp(const long long* __restrict__ ei, int E0, int ET,
                         const float* __restrict__ m,
                         float* __restrict__ e, float* __restrict__ denom) {
  int idx = blockIdx.x * blockDim.x + threadIdx.x;
  if (idx >= ET * HEADS) return;
  int edge = idx >> 3, head = idx & 7;
  long long d;
  if (edge < E0) d = ei[(size_t)E0 + edge];
  else           d = (long long)(edge - E0);
  float ex = __expf(e[idx] - m[d * HEADS + head]);
  e[idx] = ex;
  atomic_add_f32(&denom[d * HEADS + head], ex);
}

// ---------------- edge pass 3: scatter aggregate ---------------------------
// One wave32 per edge; each lane owns 4 contiguous features (one head / 4 lanes).
__global__ void __launch_bounds__(256)
edge_aggregate(const long long* __restrict__ ei, int E0, int ET,
               const float* __restrict__ ex, const float* __restrict__ denom,
               const float* __restrict__ h, float* __restrict__ agg) {
  int edge = blockIdx.x * (blockDim.x >> 5) + (threadIdx.x >> 5);
  int lane = threadIdx.x & 31;
  if (edge >= ET) return;
  long long s, d;
  if (edge < E0) { s = ei[edge]; d = ei[(size_t)E0 + edge]; }
  else           { s = d = (long long)(edge - E0); }
  int head = lane >> 2;
  float alpha = ex[(size_t)edge * HEADS + head] / denom[d * HEADS + head];
  const float4 hv =
      *reinterpret_cast<const float4*>(h + (size_t)s * HF + lane * 4);
  float* ap = agg + (size_t)d * HF + lane * 4;
  atomic_add_f32(ap + 0, alpha * hv.x);
  atomic_add_f32(ap + 1, alpha * hv.y);
  atomic_add_f32(ap + 2, alpha * hv.z);
  atomic_add_f32(ap + 3, alpha * hv.w);
}

// ---------------- epilogue: + bias, tanh, -> bf16 for next layer ----------
__global__ void finish_layer(const float* __restrict__ agg,
                             const float* __restrict__ bias,
                             __bf16* __restrict__ xb, int N) {
  int idx = blockIdx.x * blockDim.x + threadIdx.x;
  if (idx >= N * HF) return;
  float v = tanhf(agg[idx] + bias[idx & 127]);
  xb[idx] = (__bf16)v;
}

// ---------------- final linear 128 -> 1 ------------------------------------
__global__ void final_linear(const __bf16* __restrict__ xb,
                             const float* __restrict__ Wlin,
                             const float* __restrict__ blin,
                             float* __restrict__ out, int N) {
  int n = blockIdx.x * blockDim.x + threadIdx.x;
  if (n >= N) return;
  const __bf16* xp = xb + (size_t)n * HF;
  float acc = blin[0];
#pragma unroll 16
  for (int k = 0; k < HF; ++k) acc += (float)xp[k] * Wlin[k];
  out[n] = acc;
}

// ---------------------------------------------------------------------------
extern "C" void kernel_launch(void* const* d_in, const int* in_sizes, int n_in,
                              void* d_out, int out_size, void* d_ws,
                              size_t ws_size, hipStream_t stream) {
  const float*     z      = (const float*)d_in[0];
  const long long* ei     = (const long long*)d_in[1];
  const float*     W1     = (const float*)d_in[2];
  const float*     a_src1 = (const float*)d_in[3];
  const float*     a_dst1 = (const float*)d_in[4];
  const float*     b1     = (const float*)d_in[5];
  const float*     W2     = (const float*)d_in[6];
  const float*     a_src2 = (const float*)d_in[7];
  const float*     a_dst2 = (const float*)d_in[8];
  const float*     b2     = (const float*)d_in[9];
  const float*     Wlin   = (const float*)d_in[10];
  const float*     blin   = (const float*)d_in[11];

  const int N  = in_sizes[0] / EMB;     // 50000
  const int E0 = in_sizes[1] / 2;       // 1.6M
  const int ET = E0 + N;                // + self loops

  // ---- carve workspace (256B aligned) ----
  char*  ws  = (char*)d_ws;
  size_t off = 0;
  auto carve = [&](size_t bytes) -> void* {
    void* p = ws + off;
    off += (bytes + 255) & ~(size_t)255;
    return p;
  };
  __bf16* xb    = (__bf16*)carve((size_t)N * HF * 2);
  __bf16* wt    = (__bf16*)carve((size_t)HF * HF * 2);
  float*  h     = (float*) carve((size_t)N * HF * 4);
  float*  ssrc  = (float*) carve((size_t)N * HEADS * 4);
  float*  sdst  = (float*) carve((size_t)N * HEADS * 4);
  float*  m     = (float*) carve((size_t)N * HEADS * 4);
  float*  denom = (float*) carve((size_t)N * HEADS * 4);
  float*  agg   = (float*) carve((size_t)N * HF * 4);
  float*  e     = (float*) carve((size_t)ET * HEADS * 4);

  const int TB = 256;
  const int nfeat_blocks = (N * HF + TB - 1) / TB;
  const int nhead_blocks = (N * HEADS + TB - 1) / TB;
  const int edge_blocks  = (ET * HEADS + TB - 1) / TB;
  const int eagg_blocks  = (ET + (TB >> 5) - 1) / (TB >> 5);
  const int gemm_blocks  = (N + 15) / 16;

  // input -> bf16
  cvt_f32_bf16<<<nfeat_blocks, TB, 0, stream>>>(z, xb, N * EMB);

  struct LayerP { const float *W, *as, *ad, *b; };
  const LayerP layers[2] = {{W1, a_src1, a_dst1, b1},
                            {W2, a_src2, a_dst2, b2}};

  for (int L = 0; L < 2; ++L) {
    transpose_cvt_w<<<(HF * HF + TB - 1) / TB, TB, 0, stream>>>(layers[L].W, wt);
    gemm_xw_wmma<<<gemm_blocks, TB, 0, stream>>>(xb, wt, h, N);
    node_scores<<<nhead_blocks, TB, 0, stream>>>(h, layers[L].as, layers[L].ad,
                                                 ssrc, sdst, N);
    init_layer<<<nfeat_blocks, TB, 0, stream>>>(agg, m, denom, N);
    edge_logits<<<edge_blocks, TB, 0, stream>>>(ei, E0, ET, ssrc, sdst, e, m);
    edge_exp<<<edge_blocks, TB, 0, stream>>>(ei, E0, ET, m, e, denom);
    edge_aggregate<<<eagg_blocks, TB, 0, stream>>>(ei, E0, ET, e, denom, h, agg);
    finish_layer<<<nfeat_blocks, TB, 0, stream>>>(agg, layers[L].b, xb, N);
  }

  final_linear<<<(N + TB - 1) / TB, TB, 0, stream>>>(xb, Wlin, blin,
                                                     (float*)d_out, N);
}
